// BEV_UNet_30880814858722
// MI455X (gfx1250) — matrix operand order repaired and checked by reading, not compiled, not run on hardware
//
#include <hip/hip_runtime.h>

// ---------------------------------------------------------------------------
// Types for CDNA5 WMMA (wave32): V_WMMA_F32_16X16X32_F16
// ---------------------------------------------------------------------------
typedef _Float16 h8   __attribute__((ext_vector_type(8)));
typedef _Float16 v16h __attribute__((ext_vector_type(16)));
typedef float    v8f  __attribute__((ext_vector_type(8)));

#define P_TOTAL  240000          // B*N = 2*120000
#define CELLS    131072          // 2*256*256
#define EPS_BN   1e-5f
#define NEG_MAX  (-3.402823466e+38f)

// f32 atomic max via CDNA5 global_atomic_max_num_f32 (no-return -> STOREcnt)
__device__ inline void atomic_max_f32(float* p, float v) {
    asm volatile("global_atomic_max_num_f32 %0, %1, off" : : "v"(p), "v"(v) : "memory");
}

// ---------------------------------------------------------------------------
// Init: pooled = -FLT_MAX, cnt = 0, stats = 0
// ---------------------------------------------------------------------------
__global__ void init_kernel(float* pooled, unsigned* cnt, float* st) {
    long i = (long)blockIdx.x * blockDim.x + threadIdx.x;
    long np = (long)CELLS * 512;
    long stride = (long)gridDim.x * blockDim.x;
    for (long k = i; k < np; k += stride) pooled[k] = NEG_MAX;
    if (i < CELLS) cnt[i] = 0u;
    if (i < 1024)  st[i]  = 0.f;
}

__global__ void zero_stats(float* st) {
    int i = blockIdx.x * 256 + threadIdx.x;
    if (i < 1024) st[i] = 0.f;
}

// ---------------------------------------------------------------------------
// Column statistics: sum -> st[0..F), sumsq -> st[512..512+F)
// ---------------------------------------------------------------------------
template <typename T, int F>
__global__ void colstats(const T* __restrict__ X, float* __restrict__ st, int P) {
    __shared__ float ssum[256], ssq[256];
    int lane = threadIdx.x & 31;   // feature within 32-chunk
    int sub  = threadIdx.x >> 5;   // row subgroup 0..7
    int f    = blockIdx.x * 32 + lane;
    float s = 0.f, q = 0.f;
    if (f < F) {
        long base = (long)blockIdx.y * 2048;
        for (int r = sub; r < 2048; r += 8) {
            long row = base + r;
            if (row < P) {
                float v = (float)X[row * F + f];
                s += v; q += v * v;
            }
        }
    }
    ssum[threadIdx.x] = s; ssq[threadIdx.x] = q;
    __syncthreads();
    if (sub == 0) {
        for (int i = 1; i < 8; i++) { s += ssum[lane + 32 * i]; q += ssq[lane + 32 * i]; }
        if (f < F) { atomicAdd(&st[f], s); atomicAdd(&st[512 + f], q); }
    }
}

// fold BN: s = g * rsqrt(var+eps), t = be - mean*s
__global__ void fold_bn(const float* st, const float* g, const float* be,
                        float* s, float* t, int F, float invP) {
    int f = blockIdx.x * blockDim.x + threadIdx.x;
    if (f < F) {
        float m  = st[f] * invP;
        float v  = st[512 + f] * invP - m * m;
        float sc = g[f] * rsqrtf(v + EPS_BN);
        s[f] = sc;
        t[f] = be[f] - m * sc;
    }
}

// ---------------------------------------------------------------------------
// Weight prep: transpose to f16 [KOUT][KIN]; layer1 folds BN0 affine.
// ---------------------------------------------------------------------------
__global__ void prepW1(const float* W1, const float* b1, const float* s0, const float* t0,
                       _Float16* W1T, float* b1f) {
    int n = threadIdx.x;
    if (n < 64) {
        float bb = b1[n];
        for (int k = 0; k < 32; k++) {
            float w = (k < 3) ? W1[k * 64 + n] : 0.f;
            W1T[n * 32 + k] = (_Float16)((k < 3) ? w * s0[k] : 0.f);
            if (k < 3) bb += t0[k] * w;
        }
        b1f[n] = bb;
    }
}

__global__ void prepWT(const float* __restrict__ W, _Float16* __restrict__ WT,
                       int KIN, int KOUT) {
    int i = blockIdx.x * blockDim.x + threadIdx.x;
    if (i < KIN * KOUT) {
        int k = i / KOUT, n = i % KOUT;
        WT[n * KIN + k] = (_Float16)W[i];
    }
}

// ---------------------------------------------------------------------------
// cnt: mark occupied cells (idempotent store, race-benign)
// ---------------------------------------------------------------------------
__global__ void cnt_kernel(const long long* __restrict__ gind, unsigned* __restrict__ cnt, int P) {
    int p = blockIdx.x * blockDim.x + threadIdx.x;
    if (p >= P) return;
    long b  = (p < (P >> 1)) ? 0 : 1;
    long gx = (long)gind[(long)p * 2 + 0];
    long gz = (long)gind[(long)p * 2 + 1];
    cnt[b * 65536 + gx * 256 + gz] = 1u;
}

// ---------------------------------------------------------------------------
// WMMA GEMM. One wave computes a 16-row output tile over all of KOUT.
// MODE 0: A = pt_fea f32 [P,3] (BN0 folded into W1T/b1f), store Y f16
// MODE 1: A = f16 [P,KIN], affine(s,t)+ReLU at load, store Y f16
// MODE 2: same A path; epilogue scatters atomic-max into pooled (+b4)
// MODE 3: A = pooled f32 (0 if cell empty), epilogue: comp = mask*relu(.+bc)
// ---------------------------------------------------------------------------
template <int KIN, int KOUT, int MODE>
__global__ __launch_bounds__(128)
void gemm_wmma(const float* __restrict__ Xf, const _Float16* __restrict__ Xh,
               const float* __restrict__ sA, const float* __restrict__ tA,
               const _Float16* __restrict__ WT, const float* __restrict__ bias,
               _Float16* __restrict__ Y, float* __restrict__ pooled,
               const long long* __restrict__ gind, const unsigned* __restrict__ cnt,
               float* __restrict__ comp, int P) {
    constexpr int NT   = KOUT / 16;
    constexpr int NBLK = (NT < 8) ? NT : 8;
    int wave = blockIdx.x * (blockDim.x >> 5) + (threadIdx.x >> 5);
    int m0 = wave * 16;
    if (m0 >= P) return;                 // whole-wave uniform: EXEC stays all-1s
    int lane = threadIdx.x & 31;
    int l = lane & 15, hi = lane >> 4;
    int arow = m0 + l;                   // row this lane supplies for A

    long segs[8] = {};
    if constexpr (MODE == 2) {
        for (int v = 0; v < 8; v++) {
            int m = m0 + v + 8 * hi;
            long b  = (m < (P >> 1)) ? 0 : 1;
            long gx = (long)gind[(long)m * 2 + 0];
            long gz = (long)gind[(long)m * 2 + 1];
            segs[v] = b * 65536 + gx * 256 + gz;
        }
    }
    unsigned cflag = 0;
    if constexpr (MODE == 3) cflag = cnt[arow];

    for (int nb = 0; nb < NT; nb += NBLK) {
        v8f acc[NBLK];
        for (int j = 0; j < NBLK; j++) acc[j] = (v8f)(0.f);

        for (int kk = 0; kk < KIN; kk += 32) {
            // ---- build A fragment (ISA 16-bit A 16x32 layout) ----
            v16h a;
            if constexpr (MODE == 0) {
                float x0 = Xf[arow * 3 + 0];
                float x1 = Xf[arow * 3 + 1];
                float x2 = Xf[arow * 3 + 2];
                for (int e = 0; e < 16; e++) {
                    int k = (e & 7) + 16 * (e >> 3) + 8 * hi;
                    float v = (k == 0) ? x0 : (k == 1) ? x1 : (k == 2) ? x2 : 0.f;
                    a[e] = (_Float16)v;
                }
            } else if constexpr (MODE == 3) {
                const float* xr = Xf + (long)arow * KIN + kk;
                for (int e = 0; e < 16; e++) {
                    int k = (e & 7) + 16 * (e >> 3) + 8 * hi;
                    a[e] = (_Float16)(cflag ? xr[k] : 0.f);
                }
            } else {
                const _Float16* xr = Xh + (long)arow * KIN + kk;
                if (kk + 32 < KIN) __builtin_prefetch(xr + 32, 0, 1);
                h8 r0 = *(const h8*)(xr + 8 * hi);
                h8 r1 = *(const h8*)(xr + 16 + 8 * hi);
                for (int e = 0; e < 8; e++) {
                    int k0 = kk + 8 * hi + e;
                    int k1 = kk + 16 + 8 * hi + e;
                    float v0 = (float)r0[e] * sA[k0] + tA[k0];
                    float v1 = (float)r1[e] * sA[k1] + tA[k1];
                    a[e]     = (_Float16)fmaxf(v0, 0.f);
                    a[e + 8] = (_Float16)fmaxf(v1, 0.f);
                }
            }
            // ---- B fragments + WMMA (A reused across NBLK n-tiles) ----
            for (int j = 0; j < NBLK; j++) {
                int n0 = (nb + j) * 16;
                const _Float16* wr = WT + (long)(n0 + l) * KIN + kk + 8 * hi;
                h8 w0 = *(const h8*)wr;
                h8 w1 = *(const h8*)(wr + 16);
                v16h b;
                for (int e = 0; e < 8; e++) { b[e] = w0[e]; b[e + 8] = w1[e]; }
                acc[j] = __builtin_amdgcn_wmma_f32_16x16x32_f16(
                    false, a, false, b, (short)0, acc[j], false, false);
            }
        }

        // ---- epilogue ----
        for (int j = 0; j < NBLK; j++) {
            int n = (nb + j) * 16 + l;
            float bv = bias ? bias[n] : 0.f;
            if constexpr (MODE == 2) {
                for (int v = 0; v < 8; v++) {
                    float hval = acc[j][v] + bv;
                    atomic_max_f32(pooled + segs[v] * 512 + n, hval);
                }
            } else if constexpr (MODE == 3) {
                for (int v = 0; v < 8; v++) {
                    int m = m0 + v + 8 * hi;
                    float o = cnt[m] ? fmaxf(acc[j][v] + bv, 0.f) : 0.f;
                    comp[(long)m * KOUT + n] = o;
                }
            } else {
                for (int v = 0; v < 8; v++) {
                    int m = m0 + v + 8 * hi;
                    Y[(long)m * KOUT + n] = (_Float16)(acc[j][v] + bv);
                }
            }
        }
    }
}

// ---------------------------------------------------------------------------
// Output: channels 0..31 = occupancy transpose, 32..63 = 3x3 maxpool of comp
// out[b][c][x][z], total 2*64*256*256
// ---------------------------------------------------------------------------
__global__ void out_kernel(const float* __restrict__ occ, const float* __restrict__ comp,
                           float* __restrict__ out) {
    int idx = blockIdx.x * blockDim.x + threadIdx.x;
    if (idx >= 2 * 64 * 256 * 256) return;
    int z = idx & 255;
    int x = (idx >> 8) & 255;
    int c = (idx >> 16) & 63;
    int b = idx >> 22;
    float r;
    if (c < 32) {
        // occupancy [B,1,X,H,Z]: occu[b,h,x,z] = occupancy[b,0,x,h,z]
        r = occ[(((long)b * 256 + x) * 32 + c) * 256 + z];
    } else {
        int ch = c - 32;
        r = NEG_MAX;
        for (int dx = -1; dx <= 1; dx++)
            for (int dz = -1; dz <= 1; dz++) {
                int xx = x + dx, zz = z + dz;
                if (xx >= 0 && xx < 256 && zz >= 0 && zz < 256) {
                    float v = comp[(((long)b * 256 + xx) * 256 + zz) * 32 + ch];
                    r = fmaxf(r, v);
                }
            }
    }
    out[idx] = r;
}

// ---------------------------------------------------------------------------
// Host launch
// ---------------------------------------------------------------------------
extern "C" void kernel_launch(void* const* d_in, const int* in_sizes, int n_in,
                              void* d_out, int out_size, void* d_ws, size_t ws_size,
                              hipStream_t stream) {
    const int P = P_TOTAL;
    const float*     pt_fea = (const float*)d_in[0];
    const long long* gind   = (const long long*)d_in[1];   // int64 in reference
    const float*     occ    = (const float*)d_in[2];
    const float* W1 = (const float*)d_in[3];  const float* b1 = (const float*)d_in[4];
    const float* W2 = (const float*)d_in[5];  const float* b2 = (const float*)d_in[6];
    const float* W3 = (const float*)d_in[7];  const float* b3 = (const float*)d_in[8];
    const float* W4 = (const float*)d_in[9];  const float* b4 = (const float*)d_in[10];
    const float* g0 = (const float*)d_in[11]; const float* be0 = (const float*)d_in[12];
    const float* g1 = (const float*)d_in[13]; const float* be1 = (const float*)d_in[14];
    const float* g2 = (const float*)d_in[15]; const float* be2 = (const float*)d_in[16];
    const float* g3 = (const float*)d_in[17]; const float* be3 = (const float*)d_in[18];
    const float* Wc = (const float*)d_in[19]; const float* bc = (const float*)d_in[20];
    float* out = (float*)d_out;

    // workspace carve-up (256B aligned)
    char* w = (char*)d_ws;
    auto alloc = [&](size_t bytes) -> char* {
        char* p = w;
        w += (bytes + 255) & ~(size_t)255;
        return p;
    };
    _Float16* y1   = (_Float16*)alloc((size_t)P * 64 * 2);
    _Float16* y2   = (_Float16*)alloc((size_t)P * 128 * 2);
    _Float16* y3   = (_Float16*)alloc((size_t)P * 256 * 2);
    float*    pooled = (float*)alloc((size_t)CELLS * 512 * 4);
    unsigned* cnt  = (unsigned*)alloc((size_t)CELLS * 4);
    float*    comp = (float*)alloc((size_t)CELLS * 32 * 4);
    float*    st   = (float*)alloc(1024 * 4);
    float* s0 = (float*)alloc(32 * 4);  float* t0 = (float*)alloc(32 * 4);
    float* s1 = (float*)alloc(64 * 4);  float* t1 = (float*)alloc(64 * 4);
    float* s2 = (float*)alloc(128 * 4); float* t2 = (float*)alloc(128 * 4);
    float* s3 = (float*)alloc(256 * 4); float* t3 = (float*)alloc(256 * 4);
    _Float16* W1T = (_Float16*)alloc(64 * 32 * 2);
    float*    b1f = (float*)alloc(64 * 4);
    _Float16* W2T = (_Float16*)alloc(128 * 64 * 2);
    _Float16* W3T = (_Float16*)alloc(256 * 128 * 2);
    _Float16* W4T = (_Float16*)alloc(512 * 256 * 2);
    _Float16* WcT = (_Float16*)alloc(32 * 512 * 2);

    const float invP = 1.0f / (float)P;
    const int rowBlocks = (P + 2047) / 2048;     // 118
    const int gemmBlocks = (P / 16) / 4;         // 3750 (exact)
    const int cellBlocks = (CELLS / 16) / 4;     // 2048 (exact)

    // 0) init pooled/cnt/stats
    init_kernel<<<8192, 256, 0, stream>>>(pooled, cnt, st);

    // 1) BN0 stats on pt_fea, fold, prep weights
    colstats<float, 3><<<dim3(1, rowBlocks), 256, 0, stream>>>(pt_fea, st, P);
    fold_bn<<<1, 256, 0, stream>>>(st, g0, be0, s0, t0, 3, invP);
    prepW1<<<1, 64, 0, stream>>>(W1, b1, s0, t0, W1T, b1f);
    prepWT<<<(64 * 128 + 255) / 256, 256, 0, stream>>>(W2, W2T, 64, 128);
    prepWT<<<(128 * 256 + 255) / 256, 256, 0, stream>>>(W3, W3T, 128, 256);
    prepWT<<<(256 * 512 + 255) / 256, 256, 0, stream>>>(W4, W4T, 256, 512);
    prepWT<<<(512 * 32 + 255) / 256, 256, 0, stream>>>(Wc, WcT, 512, 32);

    // 2) Layer 1: (BN0-folded) Linear 3->64
    gemm_wmma<32, 64, 0><<<gemmBlocks, 128, 0, stream>>>(
        pt_fea, nullptr, nullptr, nullptr, W1T, b1f, y1,
        nullptr, nullptr, nullptr, nullptr, P);

    // 3) BN1 stats, Layer 2: 64->128
    zero_stats<<<4, 256, 0, stream>>>(st);
    colstats<_Float16, 64><<<dim3(2, rowBlocks), 256, 0, stream>>>(y1, st, P);
    fold_bn<<<1, 256, 0, stream>>>(st, g1, be1, s1, t1, 64, invP);
    gemm_wmma<64, 128, 1><<<gemmBlocks, 128, 0, stream>>>(
        nullptr, y1, s1, t1, W2T, b2, y2,
        nullptr, nullptr, nullptr, nullptr, P);

    // 4) BN2 stats, Layer 3: 128->256
    zero_stats<<<4, 256, 0, stream>>>(st);
    colstats<_Float16, 128><<<dim3(4, rowBlocks), 256, 0, stream>>>(y2, st, P);
    fold_bn<<<1, 256, 0, stream>>>(st, g2, be2, s2, t2, 128, invP);
    gemm_wmma<128, 256, 1><<<gemmBlocks, 128, 0, stream>>>(
        nullptr, y2, s2, t2, W3T, b3, y3,
        nullptr, nullptr, nullptr, nullptr, P);

    // 5) BN3 stats, occupancy flags, Layer 4: 256->512 fused with segment-max scatter
    zero_stats<<<4, 256, 0, stream>>>(st);
    colstats<_Float16, 256><<<dim3(8, rowBlocks), 256, 0, stream>>>(y3, st, P);
    fold_bn<<<1, 256, 0, stream>>>(st, g3, be3, s3, t3, 256, invP);
    cnt_kernel<<<(P + 255) / 256, 256, 0, stream>>>(gind, cnt, P);
    gemm_wmma<256, 512, 2><<<gemmBlocks, 128, 0, stream>>>(
        nullptr, y3, s3, t3, W4T, b4, nullptr,
        pooled, gind, nullptr, nullptr, P);

    // 6) Compression 512->32 on cells (masked), ReLU
    gemm_wmma<512, 32, 3><<<cellBlocks, 128, 0, stream>>>(
        pooled, nullptr, nullptr, nullptr, WcT, bc, nullptr,
        nullptr, nullptr, cnt, comp, CELLS);

    // 7) 3x3 maxpool + occupancy concat
    out_kernel<<<(2 * 64 * 256 * 256) / 256, 256, 0, stream>>>(occ, comp, out);
}
